// Personalized_Adaptor_48919677501455
// MI455X (gfx1250) — matrix operand order repaired and checked by reading, not compiled, not run on hardware
//
#include <hip/hip_runtime.h>

// Personalized LoRA adaptor: out = input@W^T + (input@right)@left * 1/16
// B=4, S=2048, C=R=4096, P=1024, lora=16.
// Main GEMM uses 3xBF16 split-precision WMMA (v_wmma_f32_16x16x32_bf16),
// rank-16 correction fused into the fp32 epilogue.

#define BATCH 4
#define SEQ   2048
#define NROWS 4096
#define NCOLS 4096
#define PDIM  1024
#define LORA  16
#define MTOT  (BATCH * SEQ)          // 8192
#define GEN_SCALE   (1.0f / 16.0f)
#define ADAPT_SCALE (1.0f / 16.0f)

#define KSTEP 32
#define LDSS  40                      // LDS row stride in halves (80B, padded)
#define NKT   (NCOLS / KSTEP)         // 128 k-steps

typedef __attribute__((ext_vector_type(16))) __bf16 v16bf;
typedef __attribute__((ext_vector_type(8)))  __bf16 v8bf;
typedef __attribute__((ext_vector_type(8)))  float  v8f;
typedef __attribute__((ext_vector_type(4)))  float  v4f;

__device__ __forceinline__ unsigned short f2bf(float f) {
  union { float f; unsigned u; } v; v.f = f;
  unsigned u = v.u + 0x7FFFu + ((v.u >> 16) & 1u);   // RNE
  return (unsigned short)(u >> 16);
}
__device__ __forceinline__ float bf2f(unsigned short h) {
  union { float f; unsigned u; } v; v.u = ((unsigned)h) << 16; return v.f;
}

// ---------------- Stage 0: amat[m][b][d] = pr[b] . rw_m[:,d]  (tiny) --------
__global__ void padaptor_hyper(const float* __restrict__ pr,
                               const float* __restrict__ rwr,
                               const float* __restrict__ rwl,
                               float* __restrict__ amat /* [2][4][16] */) {
  int t = threadIdx.x;
  if (t >= 128) return;
  const float* rw = (t & 64) ? rwl : rwr;
  int b = (t >> 4) & 3;
  int d = t & 15;
  const float* prb = pr + b * PDIM;
  float acc = 0.f;
  for (int p = 0; p < PDIM; ++p) acc += prb[p] * rw[p * LORA + d];
  amat[t] = acc;
}

// ---------------- Stage 1: factor[b][j] = GEN_SCALE * amat[b] . lw[:,j] -----
__global__ void __launch_bounds__(256)
padaptor_gen(const float* __restrict__ amat /* [4][16] */,
             const float* __restrict__ lw   /* [16][65536] */,
             float* __restrict__ om         /* [4][65536] */) {
  const size_t LW = (size_t)NCOLS * LORA; // 65536
  int j = blockIdx.x * 256 + threadIdx.x;
  float a0 = 0.f, a1 = 0.f, a2 = 0.f, a3 = 0.f;
#pragma unroll
  for (int d = 0; d < LORA; ++d) {
    float w = lw[(size_t)d * LW + j];
    a0 += amat[0 * LORA + d] * w;
    a1 += amat[1 * LORA + d] * w;
    a2 += amat[2 * LORA + d] * w;
    a3 += amat[3 * LORA + d] * w;
  }
  om[0 * LW + j] = a0 * GEN_SCALE;
  om[1 * LW + j] = a1 * GEN_SCALE;
  om[2 * LW + j] = a2 * GEN_SCALE;
  om[3 * LW + j] = a3 * GEN_SCALE;
}

// ---------------- Stage 2: v1[row][d] = input[row] . right[b][:,d] ----------
__global__ void __launch_bounds__(256)
padaptor_v1(const float* __restrict__ input,
            const float* __restrict__ rightw /* [4][4096][16] */,
            float* __restrict__ v1           /* [8192][16] */) {
  __shared__ float red[256][LORA + 1];
  const int row = blockIdx.x;
  const int b = row / SEQ;
  const float* inrow = input + (size_t)row * NCOLS;
  const float* rb = rightw + (size_t)b * NCOLS * LORA;
  const int t = threadIdx.x;
  float p[LORA];
#pragma unroll
  for (int d = 0; d < LORA; ++d) p[d] = 0.f;
  for (int c = t; c < NCOLS; c += 256) {
    float x = inrow[c];
    const v4f* rr = (const v4f*)(rb + (size_t)c * LORA);
#pragma unroll
    for (int q = 0; q < 4; ++q) {
      v4f r4 = rr[q];
#pragma unroll
      for (int e = 0; e < 4; ++e) p[q * 4 + e] += x * r4[e];
    }
  }
#pragma unroll
  for (int d = 0; d < LORA; ++d) red[t][d] = p[d];
  __syncthreads();
  for (int off = 128; off > 0; off >>= 1) {
    if (t < off) {
#pragma unroll
      for (int d = 0; d < LORA; ++d) red[t][d] += red[t + off][d];
    }
    __syncthreads();
  }
  if (t < LORA) v1[(size_t)row * LORA + t] = red[0][t];
}

// ---------------- Stage 3: main GEMM + fused rank-16 epilogue ---------------
__device__ __forceinline__ void load_regs(const float* __restrict__ g, v4f* r, int tid) {
#pragma unroll
  for (int i = 0; i < 4; ++i) {
    int idx = tid + i * 256;            // float4 index within 128x32 tile
    int rr = idx >> 3, q = idx & 7;
    r[i] = *(const v4f*)(g + (size_t)rr * NCOLS + q * 4);
  }
}

__device__ __forceinline__ void store_lds(const v4f* r, unsigned short* hi,
                                          unsigned short* lo, int tid) {
#pragma unroll
  for (int i = 0; i < 4; ++i) {
    int idx = tid + i * 256;
    int rr = idx >> 3, q = idx & 7;
    unsigned long long ph = 0, pl = 0;
#pragma unroll
    for (int e = 0; e < 4; ++e) {
      float xv = r[i][e];
      unsigned short h = f2bf(xv);
      unsigned short l = f2bf(xv - bf2f(h));   // residual for 3xBF16 scheme
      ph |= (unsigned long long)h << (16 * e);
      pl |= (unsigned long long)l << (16 * e);
    }
    int o = rr * LDSS + q * 4;
    *(unsigned long long*)(hi + o) = ph;
    *(unsigned long long*)(lo + o) = pl;
  }
}

// A fragment per ISA 7.12.2 (16-bit A 16x32): lane r=lane&15, h=lane>>4 holds
// row r, K in {8h..8h+7} U {16+8h..23+8h}.
__device__ __forceinline__ v16bf frag_a(const unsigned short* tile, int mrow, int lane) {
  const int r = lane & 15, h = lane >> 4;
  const unsigned short* p = tile + (mrow + r) * LDSS;
  v8bf a0 = *(const v8bf*)(p + 8 * h);
  v8bf a1 = *(const v8bf*)(p + 16 + 8 * h);
  v16bf a;
#pragma unroll
  for (int i = 0; i < 8; ++i) { a[i] = a0[i]; a[i + 8] = a1[i]; }
  return a;
}

// B fragment (16-bit B 32x16): lane n=lane&15, h=lane>>4 holds column n,
// K in {8h..8h+7} U {16+8h..23+8h}. B columns are weight rows (K-contiguous) -> no transpose.
__device__ __forceinline__ v16bf frag_b(const unsigned short* tile, int ncol, int lane) {
  const int n = lane & 15, h = lane >> 4;
  const unsigned short* p = tile + (ncol + n) * LDSS;
  v8bf b0 = *(const v8bf*)(p + 8 * h);
  v8bf b1 = *(const v8bf*)(p + 16 + 8 * h);
  v16bf b;
#pragma unroll
  for (int i = 0; i < 8; ++i) { b[i] = b0[i]; b[i + 8] = b1[i]; }
  return b;
}

__global__ void __launch_bounds__(256)
padaptor_main(const float* __restrict__ input,
              const float* __restrict__ weight,
              const float* __restrict__ v1,
              const float* __restrict__ lleft /* [4][16][4096] */,
              float* __restrict__ out) {
  __shared__ unsigned short sAh[128 * LDSS];   // 10 KB each, 40 KB total
  __shared__ unsigned short sAl[128 * LDSS];
  __shared__ unsigned short sBh[128 * LDSS];
  __shared__ unsigned short sBl[128 * LDSS];

  const int tid  = threadIdx.x;
  const int lane = tid & 31;
  const int wid  = tid >> 5;           // 8 waves
  const int wm   = (wid & 3) * 32;     // 4 waves in M, 32 rows each
  const int wn   = (wid >> 2) * 64;    // 2 waves in N, 64 cols each
  const int m0   = blockIdx.y * 128;
  const int n0   = blockIdx.x * 128;

  v8f acc[2][4];
#pragma unroll
  for (int mi = 0; mi < 2; ++mi)
#pragma unroll
    for (int ni = 0; ni < 4; ++ni)
#pragma unroll
      for (int e = 0; e < 8; ++e) acc[mi][ni][e] = 0.f;

  const float* gA = input  + (size_t)m0 * NCOLS;
  const float* gB = weight + (size_t)n0 * NCOLS;

  v4f ra[4], rb[4];
  load_regs(gA, ra, tid);
  load_regs(gB, rb, tid);
  store_lds(ra, sAh, sAl, tid);
  store_lds(rb, sBh, sBl, tid);
  __syncthreads();

  for (int kt = 0; kt < NKT; ++kt) {
    if (kt + 1 < NKT) {                 // prefetch next K-slab into registers
      load_regs(gA + (kt + 1) * KSTEP, ra, tid);
      load_regs(gB + (kt + 1) * KSTEP, rb, tid);
    }
    v16bf ah[2], al[2];
#pragma unroll
    for (int mi = 0; mi < 2; ++mi) {
      ah[mi] = frag_a(sAh, wm + 16 * mi, lane);
      al[mi] = frag_a(sAl, wm + 16 * mi, lane);
    }
#pragma unroll
    for (int ni = 0; ni < 4; ++ni) {
      v16bf bh = frag_b(sBh, wn + 16 * ni, lane);
      v16bf bl = frag_b(sBl, wn + 16 * ni, lane);
#pragma unroll
      for (int mi = 0; mi < 2; ++mi) {
        // 3xBF16 error-compensated fp32 GEMM: hh + hl + lh, fp32 accum
        acc[mi][ni] = __builtin_amdgcn_wmma_f32_16x16x32_bf16(
            false, ah[mi], false, bh, (short)0, acc[mi][ni], false, false);
        acc[mi][ni] = __builtin_amdgcn_wmma_f32_16x16x32_bf16(
            false, al[mi], false, bh, (short)0, acc[mi][ni], false, false);
        acc[mi][ni] = __builtin_amdgcn_wmma_f32_16x16x32_bf16(
            false, ah[mi], false, bl, (short)0, acc[mi][ni], false, false);
      }
    }
    __syncthreads();
    if (kt + 1 < NKT) {
      store_lds(ra, sAh, sAl, tid);
      store_lds(rb, sBh, sBl, tid);
      __syncthreads();
    }
  }

  // Fused rank-16 LoRA correction, full fp32.
  // D layout: lane (n=lane&15, h=lane>>4) holds D[8h+j][n] in acc[..][j].
  const int b = m0 / SEQ;
  const float* lf = lleft + (size_t)b * LORA * NROWS;
  const int n = lane & 15, h = lane >> 4;

  float lcol[4][LORA];
#pragma unroll
  for (int ni = 0; ni < 4; ++ni) {
    int col = n0 + wn + ni * 16 + n;
#pragma unroll
    for (int d = 0; d < LORA; ++d) lcol[ni][d] = lf[(size_t)d * NROWS + col];
  }
#pragma unroll
  for (int mi = 0; mi < 2; ++mi) {
#pragma unroll
    for (int j = 0; j < 8; ++j) {
      int row = m0 + wm + mi * 16 + 8 * h + j;
      const float* vr = v1 + (size_t)row * LORA;
      float vreg[LORA];
#pragma unroll
      for (int d = 0; d < LORA; ++d) vreg[d] = vr[d];
#pragma unroll
      for (int ni = 0; ni < 4; ++ni) {
        float corr = 0.f;
#pragma unroll
        for (int d = 0; d < LORA; ++d) corr += vreg[d] * lcol[ni][d];
        int col = n0 + wn + ni * 16 + n;
        out[(size_t)row * NROWS + col] = acc[mi][ni][j] + corr * ADAPT_SCALE;
      }
    }
  }
}

// ---------------------------------------------------------------------------
extern "C" void kernel_launch(void* const* d_in, const int* in_sizes, int n_in,
                              void* d_out, int out_size, void* d_ws, size_t ws_size,
                              hipStream_t stream) {
  (void)in_sizes; (void)n_in; (void)out_size; (void)ws_size;
  const float* input  = (const float*)d_in[0];
  const float* pr     = (const float*)d_in[1];
  const float* weight = (const float*)d_in[2];
  const float* rwr    = (const float*)d_in[3];
  const float* lwr    = (const float*)d_in[4];
  const float* rwl    = (const float*)d_in[5];
  const float* lwl    = (const float*)d_in[6];
  float* out = (float*)d_out;
  float* ws  = (float*)d_ws;

  const size_t FAC = (size_t)BATCH * NCOLS * LORA;  // 262144 floats
  float* amat   = ws;                 // 128 floats  ([2][4][16])
  float* rightw = ws + 256;           // [4][4096][16]
  float* leftw  = rightw + FAC;       // [4][16][4096]
  float* v1     = leftw + FAC;        // [8192][16]
  // total workspace: ~2.63 MB

  padaptor_hyper<<<1, 128, 0, stream>>>(pr, rwr, rwl, amat);
  padaptor_gen<<<(NCOLS * LORA) / 256, 256, 0, stream>>>(amat,      lwr, rightw);
  padaptor_gen<<<(NCOLS * LORA) / 256, 256, 0, stream>>>(amat + 64, lwl, leftw);
  padaptor_v1<<<MTOT, 256, 0, stream>>>(input, rightw, v1);
  padaptor_main<<<dim3(NCOLS / 128, MTOT / 128), 256, 0, stream>>>(
      input, weight, v1, leftw, out);
}
